// TransformerBlock_69088843924221
// MI455X (gfx1250) — matrix operand order repaired and checked by reading, not compiled, not run on hardware
//
#include <hip/hip_runtime.h>
#include <hip/hip_bf16.h>

// ---------------------------------------------------------------- constants
constexpr int S  = 2048;
constexpr int E  = 2048;
constexpr int H  = 32;
constexpr int G  = 8;
constexpr int D  = 64;
constexpr int F  = 5632;
constexpr int HD = H * D;         // 2048
constexpr int GD = G * D;         // 512
constexpr int NQKV = HD + 2 * GD; // 3072

typedef __bf16 bf16;
typedef __bf16 v16bf __attribute__((ext_vector_type(16)));
typedef float  v8f   __attribute__((ext_vector_type(8)));

// ---------------------------------------------------------------- WMMA frag helpers
// A-fragment (16x32, MxK): lane r(0..15)=row M, half selects K block (+8 / +24)
__device__ inline v16bf frag_a(const bf16* p, int ld, int half, int r) {
    v16bf f;
    const bf16* q = p + r * ld + half * 8;
    ((uint4*)&f)[0] = *(const uint4*)(q);       // K = half*8 + 0..7
    ((uint4*)&f)[1] = *(const uint4*)(q + 16);  // K = 16 + half*8 + 0..7
    return f;
}
// B-fragment (32x16, KxN) with B stored [n][k]: lane r = col N, half selects K block (+8 / +24)
__device__ inline v16bf frag_b(const bf16* p, int ld, int half, int r) {
    v16bf f;
    const bf16* q = p + r * ld + half * 8;
    ((uint4*)&f)[0] = *(const uint4*)(q);
    ((uint4*)&f)[1] = *(const uint4*)(q + 16);
    return f;
}
__device__ inline v8f wmma_bf16(v16bf a, v16bf b, v8f c) {
    return __builtin_amdgcn_wmma_f32_16x16x32_bf16(false, a, false, b, (short)0, c, false, false);
}

// ---------------------------------------------------------------- CDNA5 async global->LDS
// GLOBAL_LOAD_ASYNC_TO_LDS_B128, GVS mode: LDS byte-offset in VGPR (vdst),
// 32-bit global byte-offset in VGPR (vaddr), uniform 64-bit base in SGPR pair.
// Tracked by ASYNCcnt; completed with s_wait_asynccnt.
__device__ inline void async_ld16(const bf16* lds_ptr, const bf16* gbase, unsigned byte_off) {
    unsigned lds_off = (unsigned)(size_t)lds_ptr;  // low 32 bits of generic addr = LDS offset
    asm volatile("global_load_async_to_lds_b128 %0, %1, %2"
                 :: "v"(lds_off), "v"(byte_off), "s"(gbase)
                 : "memory");
}
__device__ inline void wait_async0() {
    asm volatile("s_wait_asynccnt 0" ::: "memory");
}

// ---------------------------------------------------------------- weight convert+transpose
// in: f32 [K][N] row-major  ->  out: bf16 [N][K]
__global__ void conv_transpose(const float* __restrict__ in, bf16* __restrict__ out,
                               int K, int N) {
    size_t idx = (size_t)blockIdx.x * 256 + threadIdx.x;
    if (idx >= (size_t)K * N) return;
    size_t k = idx / N, n = idx % N;
    out[n * (size_t)K + k] = (bf16)in[idx];
}

// ---------------------------------------------------------------- RMSNorm (f32 in, bf16 out)
__global__ __launch_bounds__(256) void rmsnorm_kernel(const float* __restrict__ x,
                                                      const float* __restrict__ g,
                                                      bf16* __restrict__ out) {
    __shared__ float red[256];
    const int row = blockIdx.x;
    const float* xr = x + (size_t)row * E;
    float ss = 0.f;
    for (int i = threadIdx.x; i < E; i += 256) { float v = xr[i]; ss += v * v; }
    red[threadIdx.x] = ss;
    __syncthreads();
    for (int s = 128; s > 0; s >>= 1) {
        if ((int)threadIdx.x < s) red[threadIdx.x] += red[threadIdx.x + s];
        __syncthreads();
    }
    float inv = rsqrtf(red[0] / (float)E + 1e-8f);
    for (int i = threadIdx.x; i < E; i += 256)
        out[(size_t)row * E + i] = (bf16)(xr[i] * inv * g[i]);
}

// ---------------------------------------------------------------- RoPE + pack q/k/v
// qkv f32 [S][NQKV]; q -> qb [H][S][D] (roped); k -> kb [G][S][D] (roped);
// v -> vt [G][D][S] (transposed for PV B-fragments)
__global__ void rope_pack(const float* __restrict__ qkv,
                          const float* __restrict__ cosb, const float* __restrict__ sinb,
                          bf16* __restrict__ qb, bf16* __restrict__ kb, bf16* __restrict__ vt) {
    int idx  = blockIdx.x * 256 + threadIdx.x;      // S * (H+2G) * D threads
    int per  = (H + 2 * G) * D;
    int s    = idx / per;
    int rem  = idx % per;
    const float* rowp = qkv + (size_t)s * NQKV;
    if (rem < HD) {
        int hh = rem / D, d = rem % D;
        float v = rowp[hh * D + d];
        float o = (d < D / 2) ? -rowp[hh * D + d + D / 2] : rowp[hh * D + d - D / 2];
        qb[((size_t)hh * S + s) * D + d] = (bf16)(v * cosb[s * D + d] + o * sinb[s * D + d]);
    } else if (rem < HD + GD) {
        int t = rem - HD; int gg = t / D, d = t % D;
        float v = rowp[HD + gg * D + d];
        float o = (d < D / 2) ? -rowp[HD + gg * D + d + D / 2] : rowp[HD + gg * D + d - D / 2];
        kb[((size_t)gg * S + s) * D + d] = (bf16)(v * cosb[s * D + d] + o * sinb[s * D + d]);
    } else {
        int t = rem - HD - GD; int gg = t / D, d = t % D;
        vt[((size_t)gg * D + d) * S + s] = (bf16)rowp[HD + GD + gg * D + d];
    }
}

// ---------------------------------------------------------------- generic WMMA GEMM
// C[M][N] = A[M][K] @ B[K][N]; B given transposed bf16 [N][K].
// Double-buffered LDS, async global->LDS tile streaming.
// EPI 0: Cf = acc   1: Cf = acc + resid   2: Cb = acc   3: dual-B SwiGLU -> Cb
template <int EPI>
__global__ __launch_bounds__(256) void gemm_wmma(const bf16* __restrict__ A,
                                                 const bf16* __restrict__ Bt,
                                                 const bf16* __restrict__ Bt2,
                                                 const float* __restrict__ resid,
                                                 float* __restrict__ Cf,
                                                 bf16* __restrict__ Cb,
                                                 int M, int N, int K) {
    __shared__ bf16 As[2][128 * 32];
    __shared__ bf16 Bs[2][128 * 32];
    __shared__ bf16 Bs2[(EPI == 3) ? 2 * 128 * 32 : 32];

    const int tid  = threadIdx.x;
    const int lane = tid & 31;
    const int wid  = tid >> 5;
    const int wm   = wid >> 1;   // 0..3 (M)
    const int wn   = wid & 1;    // 0..1 (N)
    const int half = lane >> 4;
    const int r    = lane & 15;
    const int m0   = blockIdx.y * 128;
    const int n0   = blockIdx.x * 128;

    v8f acc[2][4], acc2[2][4];
    for (int i = 0; i < 2; i++)
        for (int j = 0; j < 4; j++)
            for (int e = 0; e < 8; e++) { acc[i][j][e] = 0.f; acc2[i][j][e] = 0.f; }

    auto issue_tile = [&](int k0, int buf) {
#pragma unroll
        for (int j = 0; j < 2; j++) {
            int s   = tid + j * 256;           // 512 16B-slots for a 128x32 bf16 tile
            int row = s >> 2, ch = (s & 3) * 8;
            async_ld16(&As[buf][row * 32 + ch], A,  (unsigned)(((m0 + row) * K + k0 + ch) * 2));
            async_ld16(&Bs[buf][row * 32 + ch], Bt, (unsigned)(((n0 + row) * K + k0 + ch) * 2));
            if constexpr (EPI == 3)
                async_ld16(&Bs2[buf * 4096 + row * 32 + ch], Bt2,
                           (unsigned)(((n0 + row) * K + k0 + ch) * 2));
        }
    };

    int buf = 0;
    issue_tile(0, 0);
    wait_async0();
    __syncthreads();

    for (int k0 = 0; k0 < K; k0 += 32) {
        if (k0 + 32 < K) issue_tile(k0 + 32, buf ^ 1);   // stream next tile during compute

        v16bf af[2], bfr[4], bfr2[4];
#pragma unroll
        for (int i = 0; i < 2; i++)
            af[i] = frag_a(&As[buf][(wm * 32 + i * 16) * 32], 32, half, r);
#pragma unroll
        for (int j = 0; j < 4; j++)
            bfr[j] = frag_b(&Bs[buf][(wn * 64 + j * 16) * 32], 32, half, r);
        if constexpr (EPI == 3) {
#pragma unroll
            for (int j = 0; j < 4; j++)
                bfr2[j] = frag_b(&Bs2[buf * 4096 + (wn * 64 + j * 16) * 32], 32, half, r);
        }
#pragma unroll
        for (int i = 0; i < 2; i++)
#pragma unroll
            for (int j = 0; j < 4; j++) {
                acc[i][j] = wmma_bf16(af[i], bfr[j], acc[i][j]);
                if constexpr (EPI == 3) acc2[i][j] = wmma_bf16(af[i], bfr2[j], acc2[i][j]);
            }

        wait_async0();
        __syncthreads();
        buf ^= 1;
    }

#pragma unroll
    for (int i = 0; i < 2; i++)
#pragma unroll
        for (int j = 0; j < 4; j++) {
            int col = n0 + wn * 64 + j * 16 + r;
#pragma unroll
            for (int e = 0; e < 8; e++) {
                int row = m0 + wm * 32 + i * 16 + e + 8 * half;
                size_t idx = (size_t)row * N + col;
                float v = acc[i][j][e];
                if constexpr (EPI == 0) Cf[idx] = v;
                else if constexpr (EPI == 1) Cf[idx] = v + resid[idx];
                else if constexpr (EPI == 2) Cb[idx] = (bf16)v;
                else {
                    float gate = v / (1.f + __expf(-v));       // silu
                    Cb[idx] = (bf16)(gate * acc2[i][j][e]);
                }
            }
        }
}

// ---------------------------------------------------------------- flash attention (causal, GQA)
// qb [H][S][D], kb [G][S][D], vt [G][D][S] -> outb [S][H*D] (bf16)
__global__ __launch_bounds__(128) void attn_kernel(const bf16* __restrict__ qb,
                                                   const bf16* __restrict__ kb,
                                                   const bf16* __restrict__ vt,
                                                   bf16* __restrict__ outb) {
    constexpr int LDK = 72;                 // padded LDS row (144B, 16B aligned)
    __shared__ bf16 Ks[2][64 * LDK];        // [key][d]
    __shared__ bf16 Vs[2][64 * LDK];        // [d][key]
    __shared__ bf16 Ps[4 * 16 * 64];        // per-wave P staging

    const int tid  = threadIdx.x;
    const int lane = tid & 31;
    const int w    = tid >> 5;              // 4 waves, 16 rows each
    const int half = lane >> 4;
    const int r    = lane & 15;
    const int h    = blockIdx.y;
    const int g    = h >> 2;                // H/G = 4
    const int r0   = blockIdx.x * 64;
    const int rw   = r0 + w * 16;

    const bf16* qptr = qb + ((size_t)h * S + rw) * D;
    v16bf qf0 = frag_a(qptr,      D, half, r);
    v16bf qf1 = frag_a(qptr + 32, D, half, r);

    v8f o[4];
    float mi[8], li[8];
#pragma unroll
    for (int nt = 0; nt < 4; nt++)
        for (int e = 0; e < 8; e++) o[nt][e] = 0.f;
#pragma unroll
    for (int e = 0; e < 8; e++) { mi[e] = -1e30f; li[e] = 0.f; }

    auto issue_tile = [&](int t0, int buf) {
#pragma unroll
        for (int j = 0; j < 4; j++) {       // 512 16B-slots across 128 threads
            int s = tid + j * 128;
            int row = s >> 3, ch = (s & 7) * 8;
            async_ld16(&Ks[buf][row * LDK + ch], kb,
                       (unsigned)(((g * S + t0 + row) * D + ch) * 2));
            async_ld16(&Vs[buf][row * LDK + ch], vt,
                       (unsigned)(((g * D + row) * S + t0 + ch) * 2));
        }
    };

    const int tmax = r0 + 64;
    int buf = 0;
    issue_tile(0, 0);
    wait_async0();
    __syncthreads();

    for (int t0 = 0; t0 < tmax; t0 += 64) {
        if (t0 + 64 < tmax) issue_tile(t0 + 64, buf ^ 1);

        // preload all K fragments, then run the WMMA chain back-to-back
        v16bf kf[4][2];
#pragma unroll
        for (int nt = 0; nt < 4; nt++) {
            kf[nt][0] = frag_b(&Ks[buf][nt * 16 * LDK +  0], LDK, half, r);
            kf[nt][1] = frag_b(&Ks[buf][nt * 16 * LDK + 32], LDK, half, r);
        }
        v8f sa[4];
#pragma unroll
        for (int nt = 0; nt < 4; nt++) {
            for (int e = 0; e < 8; e++) sa[nt][e] = 0.f;
            sa[nt] = wmma_bf16(qf0, kf[nt][0], sa[nt]);
            sa[nt] = wmma_bf16(qf1, kf[nt][1], sa[nt]);
        }

        // online softmax
        float p[4][8], mx[8], ls[8];
#pragma unroll
        for (int e = 0; e < 8; e++) mx[e] = -1e30f;
#pragma unroll
        for (int nt = 0; nt < 4; nt++)
            for (int e = 0; e < 8; e++) {
                int row = rw + e + 8 * half;
                int col = t0 + nt * 16 + r;
                float v = sa[nt][e] * 0.125f;          // 1/sqrt(64)
                if (col > row) v = -1e30f;             // causal mask
                p[nt][e] = v;
                mx[e] = fmaxf(mx[e], v);
            }
#pragma unroll
        for (int mk = 8; mk >= 1; mk >>= 1)
            for (int e = 0; e < 8; e++) mx[e] = fmaxf(mx[e], __shfl_xor(mx[e], mk, 32));
        float corr[8];
#pragma unroll
        for (int e = 0; e < 8; e++) {
            float mn = fmaxf(mi[e], mx[e]);
            corr[e] = __expf(mi[e] - mn);
            mi[e] = mn;
        }
#pragma unroll
        for (int nt = 0; nt < 4; nt++)
            for (int e = 0; e < 8; e++) p[nt][e] = __expf(p[nt][e] - mi[e]);
#pragma unroll
        for (int e = 0; e < 8; e++) ls[e] = p[0][e] + p[1][e] + p[2][e] + p[3][e];
#pragma unroll
        for (int mk = 8; mk >= 1; mk >>= 1)
            for (int e = 0; e < 8; e++) ls[e] += __shfl_xor(ls[e], mk, 32);
#pragma unroll
        for (int e = 0; e < 8; e++) li[e] = li[e] * corr[e] + ls[e];
#pragma unroll
        for (int nt = 0; nt < 4; nt++)
            for (int e = 0; e < 8; e++) o[nt][e] *= corr[e];

        // stage P (C-layout -> row-major) then reload as A-fragments
        bf16* Pw = &Ps[w * 16 * 64];
#pragma unroll
        for (int nt = 0; nt < 4; nt++)
            for (int e = 0; e < 8; e++)
                Pw[(e + 8 * half) * 64 + nt * 16 + r] = (bf16)p[nt][e];
        v16bf pf0 = frag_a(Pw,      64, half, r);
        v16bf pf1 = frag_a(Pw + 32, 64, half, r);

        v16bf vf[4][2];
#pragma unroll
        for (int nt = 0; nt < 4; nt++) {
            vf[nt][0] = frag_b(&Vs[buf][nt * 16 * LDK +  0], LDK, half, r);
            vf[nt][1] = frag_b(&Vs[buf][nt * 16 * LDK + 32], LDK, half, r);
        }
#pragma unroll
        for (int nt = 0; nt < 4; nt++) {
            o[nt] = wmma_bf16(pf0, vf[nt][0], o[nt]);
            o[nt] = wmma_bf16(pf1, vf[nt][1], o[nt]);
        }

        wait_async0();
        __syncthreads();
        buf ^= 1;
    }

#pragma unroll
    for (int nt = 0; nt < 4; nt++)
        for (int e = 0; e < 8; e++) {
            int row = rw + e + 8 * half;
            int col = nt * 16 + r;
            outb[(size_t)row * HD + h * D + col] = (bf16)(o[nt][e] / li[e]);
        }
}

// ---------------------------------------------------------------- launch
extern "C" void kernel_launch(void* const* d_in, const int* in_sizes, int n_in,
                              void* d_out, int out_size, void* d_ws, size_t ws_size,
                              hipStream_t stream) {
    (void)in_sizes; (void)n_in; (void)out_size; (void)ws_size;

    const float* x    = (const float*)d_in[0];
    // d_in[1] = mask (bool) — causal mask applied analytically
    const float* cosb = (const float*)d_in[2];
    const float* sinb = (const float*)d_in[3];
    const float* g1   = (const float*)d_in[4];
    const float* g2   = (const float*)d_in[5];
    const float* Wq   = (const float*)d_in[6];
    const float* Wk   = (const float*)d_in[7];
    const float* Wv   = (const float*)d_in[8];
    const float* Wo   = (const float*)d_in[9];
    const float* W1   = (const float*)d_in[10];
    const float* W2   = (const float*)d_in[11];
    const float* W3   = (const float*)d_in[12];
    float* out        = (float*)d_out;

    // workspace layout (all sizes are multiples of 256B)
    char* w = (char*)d_ws;
    size_t off = 0;
    auto take = [&](size_t bytes) { char* p = w + off; off += bytes; return p; };
    bf16*  hb     = (bf16*)take((size_t)S * E * 2);
    bf16*  wqkvt  = (bf16*)take((size_t)NQKV * E * 2);
    bf16*  wot    = (bf16*)take((size_t)E * HD * 2);
    bf16*  w1t    = (bf16*)take((size_t)F * E * 2);
    bf16*  w2t    = (bf16*)take((size_t)F * E * 2);
    bf16*  w3t    = (bf16*)take((size_t)E * F * 2);
    float* qkv    = (float*)take((size_t)S * NQKV * 4);
    bf16*  qb     = (bf16*)take((size_t)H * S * D * 2);
    bf16*  kb     = (bf16*)take((size_t)G * S * D * 2);
    bf16*  vt     = (bf16*)take((size_t)G * S * D * 2);
    bf16*  attnb  = (bf16*)take((size_t)S * HD * 2);
    float* skip   = (float*)take((size_t)S * E * 4);
    bf16*  h2b    = (bf16*)take((size_t)S * E * 2);
    bf16*  ffnu   = (bf16*)take((size_t)S * F * 2);

    auto blocks = [](size_t n) { return (unsigned)((n + 255) / 256); };

    // 1) weights -> bf16 [N][K]
    conv_transpose<<<blocks((size_t)E * HD), 256, 0, stream>>>(Wq, wqkvt,                      E, HD);
    conv_transpose<<<blocks((size_t)E * GD), 256, 0, stream>>>(Wk, wqkvt + (size_t)HD * E,     E, GD);
    conv_transpose<<<blocks((size_t)E * GD), 256, 0, stream>>>(Wv, wqkvt + (size_t)(HD+GD)*E,  E, GD);
    conv_transpose<<<blocks((size_t)HD * E), 256, 0, stream>>>(Wo, wot, HD, E);
    conv_transpose<<<blocks((size_t)E * F),  256, 0, stream>>>(W1, w1t, E, F);
    conv_transpose<<<blocks((size_t)E * F),  256, 0, stream>>>(W2, w2t, E, F);
    conv_transpose<<<blocks((size_t)F * E),  256, 0, stream>>>(W3, w3t, F, E);

    // 2) h = rmsnorm(x, g1)
    rmsnorm_kernel<<<S, 256, 0, stream>>>(x, g1, hb);

    // 3) qkv = h @ [Wq|Wk|Wv]
    gemm_wmma<0><<<dim3(NQKV / 128, S / 128), 256, 0, stream>>>(hb, wqkvt, nullptr, nullptr,
                                                                qkv, nullptr, S, NQKV, E);
    // 4) rope + pack
    rope_pack<<<blocks((size_t)S * (H + 2 * G) * D), 256, 0, stream>>>(qkv, cosb, sinb, qb, kb, vt);

    // 5) attention
    attn_kernel<<<dim3(S / 64, H), 128, 0, stream>>>(qb, kb, vt, attnb);

    // 6) skipped = x + attn @ Wo
    gemm_wmma<1><<<dim3(E / 128, S / 128), 256, 0, stream>>>(attnb, wot, nullptr, x,
                                                             skip, nullptr, S, E, HD);
    // 7) h2 = rmsnorm(skipped, g2)
    rmsnorm_kernel<<<S, 256, 0, stream>>>(skip, g2, h2b);

    // 8) u = silu(h2@W1) * (h2@W2)
    gemm_wmma<3><<<dim3(F / 128, S / 128), 256, 0, stream>>>(h2b, w1t, w2t, nullptr,
                                                             nullptr, ffnu, S, F, E);
    // 9) out = skipped + u @ W3
    gemm_wmma<1><<<dim3(E / 128, S / 128), 256, 0, stream>>>(ffnu, w3t, nullptr, skip,
                                                             out, nullptr, S, E, F);
}
